// CondModel_25795573580446
// MI455X (gfx1250) — matrix operand order repaired and checked by reading, not compiled, not run on hardware
//
#include <hip/hip_runtime.h>
#include <hip/hip_bf16.h>

#define NN   131072
#define EE   262144
#define BBG  4096
#define HID  256
#define CTX  128
#define FEAT 512
#define LAY  4

typedef __attribute__((ext_vector_type(16))) __bf16 v16bf;
typedef __attribute__((ext_vector_type(8)))  float  v8f;

union FragBF { uint4 u[2]; v16bf v; };

__device__ __forceinline__ unsigned short f2bf(float f) {
    unsigned int u = __float_as_uint(f);
    unsigned int r = (u + 0x7FFFu + ((u >> 16) & 1u)) >> 16;
    return (unsigned short)r;
}

__device__ __forceinline__ void atomicMaxF(float* addr, float val) {
    unsigned int* ua = (unsigned int*)addr;
    unsigned int old = *ua;
    while (__uint_as_float(old) < val) {
        unsigned int assumed = old;
        old = atomicCAS(ua, assumed, __float_as_uint(val));
        if (old == assumed) break;
    }
}

// CDNA5 async global->LDS copy (VFLAT GLOBAL_LOAD_ASYNC_TO_LDS_B128):
// per-lane: copies 16 bytes from global vaddr to LDS offset vdst; tracked by ASYNCcnt.
__device__ __forceinline__ void async_load_b128(unsigned lds_off, const void* gptr) {
    asm volatile("global_load_async_to_lds_b128 %0, %1, off"
                 :: "v"(lds_off), "v"(gptr) : "memory");
}
__device__ __forceinline__ void wait_async0() {
    asm volatile("s_wait_asynccnt 0" ::: "memory");
}

// ---------------- encoders ----------------
__global__ __launch_bounds__(256)
void node_enc(const float* __restrict__ nf, const float* __restrict__ Wn,
              const float* __restrict__ bn, float* __restrict__ h,
              unsigned short* __restrict__ hbf) {
    __shared__ float x[69];
    const int n = blockIdx.x, j = threadIdx.x;
    if (j < 69) x[j] = nf[(size_t)n * 69 + j];
    __syncthreads();
    float a = bn[j];
    for (int k = 0; k < 69; ++k) a += x[k] * Wn[k * HID + j];
    a = a > 0.f ? a : 0.f;
    const size_t gi = (size_t)n * HID + j;
    h[gi] = a;
    hbf[gi] = f2bf(a);
}

__global__ __launch_bounds__(256)
void edge_enc(const float* __restrict__ ef, const float* __restrict__ We,
              const float* __restrict__ be, unsigned short* __restrict__ ebf) {
    __shared__ float x[9];
    const int e = blockIdx.x, j = threadIdx.x;
    if (j < 9) x[j] = ef[(size_t)e * 9 + j];
    __syncthreads();
    float a = be[j];
    for (int k = 0; k < 9; ++k) a += x[k] * We[k * HID + j];
    a = a > 0.f ? a : 0.f;
    ebf[(size_t)e * HID + j] = f2bf(a);
}

// ---------------- weight transpose+convert: [K,N]f32 -> [N,K]bf16 ----------------
__global__ __launch_bounds__(256)
void conv_weights(const float* __restrict__ Wm, const float* __restrict__ Wu,
                  unsigned short* __restrict__ WmT, unsigned short* __restrict__ WuT) {
    const int idx = blockIdx.x * 256 + threadIdx.x;
    const int totM = LAY * 512 * HID;
    const int totU = LAY * HID * HID;
    if (idx < totM) {
        int l = idx / (512 * HID), rem = idx % (512 * HID);
        int n = rem / 512, k = rem % 512;
        WmT[idx] = f2bf(Wm[((size_t)l * 512 + k) * HID + n]);
    }
    if (idx < totU) {
        int l = idx / (HID * HID), rem = idx % (HID * HID);
        int n = rem / HID, k = rem % HID;
        WuT[idx] = f2bf(Wu[((size_t)l * HID + k) * HID + n]);
    }
}

__global__ void fill_neginf(float* __restrict__ p, int nel) {
    int i = blockIdx.x * 256 + threadIdx.x;
    if (i < nel) p[i] = -__builtin_inff();
}

// ---------------- message + scatter-add (WMMA bf16, 32 edges/block) ----------------
// m = relu([h[src] | e] @ Wm + bm), agg[dst] += m
__global__ __launch_bounds__(128)
void msg_kernel(const unsigned short* __restrict__ hbf,
                const unsigned short* __restrict__ ebf,
                const int* __restrict__ src, const int* __restrict__ dst,
                const unsigned short* __restrict__ WmT,   // [256][512] bf16
                const float* __restrict__ bm,             // [256]
                float* __restrict__ agg) {
    constexpr int KD = 512, STR = 520, MT = 32;           // 260 dwords/row (==4 mod 64)
    __shared__ __align__(16) unsigned short A[MT * STR];
    const int eb = blockIdx.x * MT;
    const int tid = threadIdx.x;

    { // async-stage A tile: 32 edges x [h[src](256) | e(256)] as bf16
        const int row = tid >> 2;                 // 0..31
        const int cb  = (tid & 3) * 128;          // 128-col slice: wholly h or wholly e
        const int e = eb + row;
        const int s = src[e];
        const unsigned short* base = (cb < 256)
            ? (hbf + (size_t)s * HID + cb)
            : (ebf + (size_t)e * HID + (cb - 256));
        const unsigned lbase = (unsigned)(size_t)&A[row * STR + cb]; // LDS offset = addr[31:0]
        #pragma unroll
        for (int c8 = 0; c8 < 16; ++c8)
            async_load_b128(lbase + c8 * 16, base + c8 * 8);
    }
    wait_async0();
    __syncthreads();

    const int lane = tid & 31;
    const int lrow = lane & 15, hk = lane >> 4;
    const int nbase = (tid >> 5) * 64;                    // 4 n-tiles per wave

    v8f acc0[4], acc1[4];
    #pragma unroll
    for (int t = 0; t < 4; ++t) {
        const float bv = bm[nbase + t * 16 + lrow];
        #pragma unroll
        for (int r = 0; r < 8; ++r) { acc0[t][r] = bv; acc1[t][r] = bv; }
    }

    FragBF bcur[4];
    #pragma unroll
    for (int t = 0; t < 4; ++t) {
        const unsigned short* bp = WmT + (size_t)(nbase + t * 16 + lrow) * KD + 16 * hk;
        bcur[t].u[0] = *(const uint4*)bp;
        bcur[t].u[1] = *(const uint4*)(bp + 8);
    }

    for (int kk = 0; kk < KD; kk += 32) {
        FragBF af0, af1;
        af0.u[0] = *(const uint4*)&A[lrow * STR + kk + 16 * hk];
        af0.u[1] = *(const uint4*)&A[lrow * STR + kk + 16 * hk + 8];
        af1.u[0] = *(const uint4*)&A[(16 + lrow) * STR + kk + 16 * hk];
        af1.u[1] = *(const uint4*)&A[(16 + lrow) * STR + kk + 16 * hk + 8];
        FragBF bnx[4];
        if (kk + 32 < KD) {
            #pragma unroll
            for (int t = 0; t < 4; ++t) {
                const unsigned short* bp =
                    WmT + (size_t)(nbase + t * 16 + lrow) * KD + (kk + 32) + 16 * hk;
                bnx[t].u[0] = *(const uint4*)bp;
                bnx[t].u[1] = *(const uint4*)(bp + 8);
            }
        }
        #pragma unroll
        for (int t = 0; t < 4; ++t) {
            acc0[t] = __builtin_amdgcn_wmma_f32_16x16x32_bf16(
                false, af0.v, false, bcur[t].v, (short)0, acc0[t], false, false);
            acc1[t] = __builtin_amdgcn_wmma_f32_16x16x32_bf16(
                false, af1.v, false, bcur[t].v, (short)0, acc1[t], false, false);
        }
        #pragma unroll
        for (int t = 0; t < 4; ++t) bcur[t] = bnx[t];
    }

    // relu + scatter-add; D elem r -> (M = r+8*hk, N = lane&15)
    #pragma unroll
    for (int mt = 0; mt < 2; ++mt) {
        int dv[8];
        #pragma unroll
        for (int r = 0; r < 8; ++r) dv[r] = dst[eb + mt * 16 + r + 8 * hk];
        #pragma unroll
        for (int t = 0; t < 4; ++t) {
            const int col = nbase + t * 16 + lrow;
            #pragma unroll
            for (int r = 0; r < 8; ++r) {
                float m = mt ? acc1[t][r] : acc0[t][r];
                if (m > 0.f) atomicAdd(&agg[(size_t)dv[r] * HID + col], m);
            }
        }
    }
}

// ---------------- update: h = LN(h + relu(agg@Wu+bu)); zeroes agg ----------------
__global__ __launch_bounds__(128)
void upd_kernel(float* __restrict__ agg,
                const unsigned short* __restrict__ WuT,   // [256][256] bf16
                const float* __restrict__ bu, const float* __restrict__ lng,
                const float* __restrict__ lnb,
                float* __restrict__ h, unsigned short* __restrict__ hbf) {
    constexpr int KD = 256, ASTR = 264, RSTR = 257, MT = 32; // 132 dw (==4 mod 64)
    __shared__ __align__(16) unsigned short A[MT * ASTR];
    __shared__ float R[MT * RSTR];
    __shared__ float mu_s[MT], rs_s[MT];
    const int nb = blockIdx.x * MT;
    const int tid = threadIdx.x;

    { // load agg tile -> bf16 LDS, zero agg for next layer's scatter
        const int row = tid >> 2;                 // 0..31
        const int cb  = (tid & 3) * 64;
        const size_t gb = (size_t)(nb + row) * KD + cb;
        float4 z; z.x = 0.f; z.y = 0.f; z.z = 0.f; z.w = 0.f;
        #pragma unroll
        for (int c4 = 0; c4 < 16; ++c4) {
            float4 v = *(const float4*)&agg[gb + c4 * 4];
            *(float4*)&agg[gb + c4 * 4] = z;
            unsigned short* pa = &A[row * ASTR + cb + c4 * 4];
            pa[0] = f2bf(v.x); pa[1] = f2bf(v.y); pa[2] = f2bf(v.z); pa[3] = f2bf(v.w);
        }
    }
    __syncthreads();

    const int lane = tid & 31;
    const int lrow = lane & 15, hk = lane >> 4;
    const int nbase = (tid >> 5) * 64;

    v8f acc0[4], acc1[4];
    #pragma unroll
    for (int t = 0; t < 4; ++t) {
        const float bv = bu[nbase + t * 16 + lrow];
        #pragma unroll
        for (int r = 0; r < 8; ++r) { acc0[t][r] = bv; acc1[t][r] = bv; }
    }
    FragBF bcur[4];
    #pragma unroll
    for (int t = 0; t < 4; ++t) {
        const unsigned short* bp = WuT + (size_t)(nbase + t * 16 + lrow) * KD + 16 * hk;
        bcur[t].u[0] = *(const uint4*)bp;
        bcur[t].u[1] = *(const uint4*)(bp + 8);
    }
    for (int kk = 0; kk < KD; kk += 32) {
        FragBF af0, af1;
        af0.u[0] = *(const uint4*)&A[lrow * ASTR + kk + 16 * hk];
        af0.u[1] = *(const uint4*)&A[lrow * ASTR + kk + 16 * hk + 8];
        af1.u[0] = *(const uint4*)&A[(16 + lrow) * ASTR + kk + 16 * hk];
        af1.u[1] = *(const uint4*)&A[(16 + lrow) * ASTR + kk + 16 * hk + 8];
        FragBF bnx[4];
        if (kk + 32 < KD) {
            #pragma unroll
            for (int t = 0; t < 4; ++t) {
                const unsigned short* bp =
                    WuT + (size_t)(nbase + t * 16 + lrow) * KD + (kk + 32) + 16 * hk;
                bnx[t].u[0] = *(const uint4*)bp;
                bnx[t].u[1] = *(const uint4*)(bp + 8);
            }
        }
        #pragma unroll
        for (int t = 0; t < 4; ++t) {
            acc0[t] = __builtin_amdgcn_wmma_f32_16x16x32_bf16(
                false, af0.v, false, bcur[t].v, (short)0, acc0[t], false, false);
            acc1[t] = __builtin_amdgcn_wmma_f32_16x16x32_bf16(
                false, af1.v, false, bcur[t].v, (short)0, acc1[t], false, false);
        }
        #pragma unroll
        for (int t = 0; t < 4; ++t) bcur[t] = bnx[t];
    }
    // residual into LDS
    #pragma unroll
    for (int mt = 0; mt < 2; ++mt) {
        #pragma unroll
        for (int t = 0; t < 4; ++t) {
            const int col = nbase + t * 16 + lrow;
            #pragma unroll
            for (int r = 0; r < 8; ++r) {
                const int M = mt * 16 + r + 8 * hk;
                float u = mt ? acc1[t][r] : acc0[t][r];
                u = u > 0.f ? u : 0.f;
                R[M * RSTR + col] = h[(size_t)(nb + M) * HID + col] + u;
            }
        }
    }
    __syncthreads();
    if (tid < MT) { // per-row LayerNorm stats
        float s = 0.f;
        for (int k = 0; k < HID; ++k) s += R[tid * RSTR + k];
        const float mu = s * (1.f / HID);
        float v = 0.f;
        for (int k = 0; k < HID; ++k) { float d = R[tid * RSTR + k] - mu; v += d * d; }
        mu_s[tid] = mu;
        rs_s[tid] = rsqrtf(v * (1.f / HID) + 1e-5f);
    }
    __syncthreads();
    {
        const int row = tid >> 2;
        const int cb  = (tid & 3) * 64;
        const float mu = mu_s[row], rs = rs_s[row];
        for (int i = 0; i < 64; ++i) {
            const int col = cb + i;
            const float v = (R[row * RSTR + col] - mu) * rs * lng[col] + lnb[col];
            const size_t gi = (size_t)(nb + row) * HID + col;
            h[gi] = v;
            hbf[gi] = f2bf(v);
        }
    }
}

// ---------------- readout: segment mean/max ----------------
__global__ __launch_bounds__(256)
void readout_kernel(const float* __restrict__ h, const int* __restrict__ b,
                    float* __restrict__ ssum, float* __restrict__ gmax,
                    float* __restrict__ cnt) {
    const int n = blockIdx.x, j = threadIdx.x;
    const int g = b[n];
    const float v = h[(size_t)n * HID + j];
    atomicAdd(&ssum[(size_t)g * HID + j], v);
    atomicMaxF(&gmax[(size_t)g * HID + j], v);
    if (j == 0) atomicAdd(&cnt[g], 1.0f);
}

// ---------------- FiLM + predictor (one block per graph) ----------------
__global__ __launch_bounds__(256)
void film_kernel(const float* __restrict__ ctx, const int* __restrict__ pid,
                 const float* __restrict__ gW1, const float* __restrict__ gb1,
                 const float* __restrict__ glng, const float* __restrict__ glnb,
                 const float* __restrict__ gW2, const float* __restrict__ gb2,
                 const float* __restrict__ bW1, const float* __restrict__ bb1,
                 const float* __restrict__ blng, const float* __restrict__ blnb,
                 const float* __restrict__ bW2, const float* __restrict__ bb2,
                 const float* __restrict__ pW1, const float* __restrict__ pb1,
                 const float* __restrict__ pW2, const float* __restrict__ pb2,
                 const float* __restrict__ ssum, const float* __restrict__ gmax,
                 const float* __restrict__ cnt, float* __restrict__ out) {
    __shared__ float c[CTX];
    __shared__ float t[HID];
    __shared__ float gam[FEAT], bet[FEAT], hmod[FEAT];
    __shared__ float red[256];
    const int g = blockIdx.x, tid = threadIdx.x;
    if (tid < CTX) c[tid] = ctx[(size_t)pid[g] * CTX + tid];
    __syncthreads();

    for (int pass = 0; pass < 2; ++pass) {
        const float* W1 = pass ? bW1 : gW1;  const float* b1 = pass ? bb1 : gb1;
        const float* lg = pass ? blng : glng; const float* lb = pass ? blnb : glnb;
        const float* W2 = pass ? bW2 : gW2;  const float* b2 = pass ? bb2 : gb2;
        float* dstv = pass ? bet : gam;
        float a = b1[tid];
        for (int k = 0; k < CTX; ++k) a += c[k] * W1[k * HID + tid];
        red[tid] = a; __syncthreads();
        for (int s = 128; s > 0; s >>= 1) { if (tid < s) red[tid] += red[tid + s]; __syncthreads(); }
        const float mu = red[0] * (1.f / HID); __syncthreads();
        const float d = a - mu;
        red[tid] = d * d; __syncthreads();
        for (int s = 128; s > 0; s >>= 1) { if (tid < s) red[tid] += red[tid + s]; __syncthreads(); }
        const float var = red[0] * (1.f / HID);
        float tn = d * rsqrtf(var + 1e-5f) * lg[tid] + lb[tid];
        t[tid] = tn > 0.f ? tn : 0.f;
        __syncthreads();
        for (int f = tid; f < FEAT; f += 256) {
            float a2 = b2[f];
            for (int k = 0; k < HID; ++k) a2 += t[k] * W2[k * FEAT + f];
            dstv[f] = a2;
        }
        __syncthreads();
    }

    { // g = [mean || max], hmod = gamma*g + beta
        float cn = cnt[g]; cn = cn > 1.f ? cn : 1.f;
        const float mv = ssum[(size_t)g * HID + tid] / cn;
        float mx = gmax[(size_t)g * HID + tid];
        if (mx == -__builtin_inff()) mx = 0.f;
        hmod[tid] = gam[tid] * mv + bet[tid];
        hmod[tid + 256] = gam[tid + 256] * mx + bet[tid + 256];
    }
    __syncthreads();
    float a = pb1[tid];
    for (int k = 0; k < FEAT; ++k) a += hmod[k] * pW1[k * HID + tid];
    a = a > 0.f ? a : 0.f;
    red[tid] = a * pW2[tid];
    __syncthreads();
    for (int s = 128; s > 0; s >>= 1) { if (tid < s) red[tid] += red[tid + s]; __syncthreads(); }
    if (tid == 0) out[g] = red[0] + pb2[0];
}

// ---------------- host launcher ----------------
extern "C" void kernel_launch(void* const* d_in, const int* in_sizes, int n_in,
                              void* d_out, int out_size, void* d_ws, size_t ws_size,
                              hipStream_t stream) {
    const float* nf   = (const float*)d_in[0];
    const int*   ei   = (const int*)  d_in[1];
    const float* ef   = (const float*)d_in[2];
    const int*   b    = (const int*)  d_in[3];
    const int*   pid  = (const int*)  d_in[4];
    const float* Wn   = (const float*)d_in[5];
    const float* bn   = (const float*)d_in[6];
    const float* We   = (const float*)d_in[7];
    const float* be   = (const float*)d_in[8];
    const float* Wm   = (const float*)d_in[9];
    const float* bm   = (const float*)d_in[10];
    const float* Wu   = (const float*)d_in[11];
    const float* bu   = (const float*)d_in[12];
    const float* lng  = (const float*)d_in[13];
    const float* lnb  = (const float*)d_in[14];
    const float* ctx  = (const float*)d_in[15];
    const float* gW1  = (const float*)d_in[16];
    const float* gb1  = (const float*)d_in[17];
    const float* glng = (const float*)d_in[18];
    const float* glnb = (const float*)d_in[19];
    const float* gW2  = (const float*)d_in[20];
    const float* gb2  = (const float*)d_in[21];
    const float* bW1  = (const float*)d_in[22];
    const float* bb1  = (const float*)d_in[23];
    const float* blng = (const float*)d_in[24];
    const float* blnb = (const float*)d_in[25];
    const float* bW2  = (const float*)d_in[26];
    const float* bb2  = (const float*)d_in[27];
    const float* pW1  = (const float*)d_in[28];
    const float* pb1  = (const float*)d_in[29];
    const float* pW2  = (const float*)d_in[30];
    const float* pb2  = (const float*)d_in[31];
    float* out = (float*)d_out;

    const int* src = ei;
    const int* dst = ei + EE;

    size_t off = 0;
    auto alloc = [&](size_t bytes) {
        void* p = (char*)d_ws + off;
        off += (bytes + 255) & ~(size_t)255;
        return p;
    };
    float*          h    = (float*)         alloc((size_t)NN * HID * 4);
    unsigned short* hbf  = (unsigned short*)alloc((size_t)NN * HID * 2);
    unsigned short* ebf  = (unsigned short*)alloc((size_t)EE * HID * 2);
    float*          agg  = (float*)         alloc((size_t)NN * HID * 4);
    float*          ssum = (float*)         alloc((size_t)BBG * HID * 4);
    float*          gmax = (float*)         alloc((size_t)BBG * HID * 4);
    float*          cnt  = (float*)         alloc((size_t)BBG * 4);
    unsigned short* WmT  = (unsigned short*)alloc((size_t)LAY * 512 * HID * 2);
    unsigned short* WuT  = (unsigned short*)alloc((size_t)LAY * HID * HID * 2);

    hipMemsetAsync(agg,  0, (size_t)NN * HID * 4, stream);
    hipMemsetAsync(ssum, 0, (size_t)BBG * HID * 4, stream);
    hipMemsetAsync(cnt,  0, (size_t)BBG * 4, stream);
    fill_neginf<<<(BBG * HID + 255) / 256, 256, 0, stream>>>(gmax, BBG * HID);
    conv_weights<<<(LAY * 512 * HID + 255) / 256, 256, 0, stream>>>(Wm, Wu, WmT, WuT);

    node_enc<<<NN, 256, 0, stream>>>(nf, Wn, bn, h, hbf);
    edge_enc<<<EE, 256, 0, stream>>>(ef, We, be, ebf);

    for (int l = 0; l < LAY; ++l) {
        msg_kernel<<<EE / 32, 128, 0, stream>>>(
            hbf, ebf, src, dst, WmT + (size_t)l * 512 * HID, bm + l * HID, agg);
        upd_kernel<<<NN / 32, 128, 0, stream>>>(
            agg, WuT + (size_t)l * HID * HID, bu + l * HID,
            lng + l * HID, lnb + l * HID, h, hbf);
    }

    readout_kernel<<<NN, 256, 0, stream>>>(h, b, ssum, gmax, cnt);
    film_kernel<<<BBG, 256, 0, stream>>>(ctx, pid, gW1, gb1, glng, glnb, gW2, gb2,
                                         bW1, bb1, blng, blnb, bW2, bb2,
                                         pW1, pb1, pW2, pb2, ssum, gmax, cnt, out);
    (void)in_sizes; (void)n_in; (void)out_size; (void)ws_size;
}
